// JustUpsample_88948772700681
// MI455X (gfx1250) — matrix operand order repaired and checked by reading, not compiled, not run on hardware
//
#include <hip/hip_runtime.h>
#include <hip/hip_bf16.h>

#define N_PTS 16384
#define K_NBR 16
#define K_TOT 32
#define CH 96
#define CH_UP 32
#define CHR 384           // CH * R
#define NOUT (N_PTS * 4)  // output rows after reshape

typedef __attribute__((ext_vector_type(16))) _Float16 v16h;
typedef __attribute__((ext_vector_type(8)))  float    v8f;

__device__ __forceinline__ v8f wmma_f16f32(v16h a, v16h b, v8f c) {
  // D(f32 16x16) = A(f16 16x32) * B(f16 32x16) + C
  return __builtin_amdgcn_wmma_f32_16x16x32_f16(false, a, false, b, (short)0, c,
                                                false, false);
}

// One-op lane 0<->16 exchange: ds_swizzle group-of-32, SWAPX16
// (offset[14:10]=xor=0x10, offset[9:5]=or=0, offset[4:0]=and=0x1f -> 0x401F)
__device__ __forceinline__ float swap_half_wave(float x) {
  return __int_as_float(__builtin_amdgcn_ds_swizzle(__float_as_int(x), 0x401f));
}

// A-matrix (16-bit, 16x32) per-lane K index for element e of the v16h:
// lanes 0-15: e0..15 -> K0..15
// lanes 16-31: e0..15 -> K16..31
__device__ __forceinline__ int a_kmap(int lane, int e) {
  return ((lane & 16) ? 16 : 0) + e;
}

// Stage a row-major f32 weight matrix W[Krows x ncols] into LDS as f16 WMMA
// B-fragments (32x16 each), fragment-major: frag f = kt*nct+ct occupies
// wfrag[f*512 .. f*512+511]; within a fragment, lane l owns halves [l*16..l*16+15]
// with K = kt*32 + (l<16 ? 0 : 16) + e, col = ct*16 + (l&15).
__device__ void stage_w_frags(_Float16* wfrag, const float* __restrict__ W,
                              int Krows, int ncols, int nct,
                              int nfrag_total, int tid, int nthreads) {
  int total = nfrag_total * 512;
  for (int n = tid; n < total; n += nthreads) {
    int f = n >> 9;
    int l = (n >> 4) & 31;
    int e = n & 15;
    int kt = f / nct, ct = f - kt * nct;
    int K = kt * 32 + ((l & 16) ? 16 : 0) + e;
    int col = ct * 16 + (l & 15);
    float v = (K < Krows) ? W[K * ncols + col] : 0.0f;
    wfrag[n] = (_Float16)v;
  }
}

// ---- gfx1250 async global->LDS path (ASYNCcnt), guarded for portability ----
#if __has_builtin(__builtin_amdgcn_global_load_async_to_lds_b32) && \
    __has_builtin(__builtin_amdgcn_s_wait_asynccnt)
#define USE_ASYNC_LDS 1
typedef __attribute__((address_space(1))) int gas_int_t;
typedef __attribute__((address_space(3))) int las_int_t;
__device__ __forceinline__ void async_g2l_b32(const float* g, float* l) {
  __builtin_amdgcn_global_load_async_to_lds_b32((gas_int_t*)g, (las_int_t*)l,
                                                0, 0);
}
#else
#define USE_ASYNC_LDS 0
#endif

// ---------------------------------------------------------------------------
// Kernel A: dense KNN, sorted top-32 per point (ascending distance).
// One thread per query point; candidate tiles double-buffered through LDS via
// gfx1250 async global->LDS loads (overlap tile fetch with the VALU scan).
// Per-thread top list lives in LDS, [slot][tid] layout (bank-conflict free).
// ---------------------------------------------------------------------------
__global__ __launch_bounds__(128) void knn_kernel(const float* __restrict__ pos,
                                                  int* __restrict__ nbr) {
  __shared__ float s_tile[2][128 * 3];
  __shared__ float s_dist[K_TOT * 128];
  __shared__ int   s_idx[K_TOT * 128];
  int tid = threadIdx.x;
  int i = blockIdx.x * 128 + tid;
  float xi = pos[i * 3 + 0], yi = pos[i * 3 + 1], zi = pos[i * 3 + 2];
#pragma unroll
  for (int s = 0; s < K_TOT; ++s) {
    s_dist[s * 128 + tid] = __builtin_inff();
    s_idx[s * 128 + tid] = 0;
  }
  float worst = __builtin_inff();

#if USE_ASYNC_LDS
  {  // prologue: tile 0 -> buffer 0
    const float* g = pos + tid * 3;
    float* l = &s_tile[0][tid * 3];
    async_g2l_b32(g + 0, l + 0);
    async_g2l_b32(g + 1, l + 1);
    async_g2l_b32(g + 2, l + 2);
  }
#endif

  for (int tile = 0; tile < N_PTS; tile += 128) {
    int buf = (tile >> 7) & 1;
#if USE_ASYNC_LDS
    __builtin_amdgcn_s_wait_asynccnt(0);
    __syncthreads();  // buf ready for all; everyone done reading buf^1
    if (tile + 128 < N_PTS) {
      const float* g = pos + (tile + 128 + tid) * 3;
      float* l = &s_tile[buf ^ 1][tid * 3];
      async_g2l_b32(g + 0, l + 0);
      async_g2l_b32(g + 1, l + 1);
      async_g2l_b32(g + 2, l + 2);
    }
#else
    __syncthreads();
    s_tile[buf][tid * 3 + 0] = pos[(tile + tid) * 3 + 0];
    s_tile[buf][tid * 3 + 1] = pos[(tile + tid) * 3 + 1];
    s_tile[buf][tid * 3 + 2] = pos[(tile + tid) * 3 + 2];
    __syncthreads();
#endif
    const float* tp = s_tile[buf];
    for (int t = 0; t < 128; ++t) {
      int j = tile + t;
      float dx = xi - tp[t * 3 + 0];
      float dy = yi - tp[t * 3 + 1];
      float dz = zi - tp[t * 3 + 2];
      float d = dx * dx + dy * dy + dz * dz;
      if (j == i || d >= worst) continue;
      int p = K_TOT - 1;
      while (p > 0 && s_dist[(p - 1) * 128 + tid] > d) {
        s_dist[p * 128 + tid] = s_dist[(p - 1) * 128 + tid];
        s_idx[p * 128 + tid]  = s_idx[(p - 1) * 128 + tid];
        --p;
      }
      s_dist[p * 128 + tid] = d;
      s_idx[p * 128 + tid]  = j;
      worst = s_dist[(K_TOT - 1) * 128 + tid];
    }
  }
#pragma unroll
  for (int s = 0; s < K_TOT; ++s) nbr[i * K_TOT + s] = s_idx[s * 128 + tid];
}

// ---------------------------------------------------------------------------
// Kernel B: EdgeConv-pre (dilation 2, 6->96, max over 16 nbrs) + bottleneck
// (96->32, leaky 0.2). One thread per point; h kept in registers (unrolled),
// spilled through LDS (f16) for the 96->32 contraction.
// ---------------------------------------------------------------------------
__global__ __launch_bounds__(256) void preconv_kernel(
    const float* __restrict__ pos, const int* __restrict__ nbr,
    const float* __restrict__ W_pre, const float* __restrict__ b_pre,
    const float* __restrict__ W_bn, const float* __restrict__ b_bn,
    float* __restrict__ xf) {
  __shared__ float s_Wp[6 * CH];
  __shared__ float s_Wbn[CH * CH_UP];
  __shared__ _Float16 s_h[CH * 256];
  int tid = threadIdx.x;
  for (int n = tid; n < 6 * CH; n += 256) s_Wp[n] = W_pre[n];
  for (int n = tid; n < CH * CH_UP; n += 256) s_Wbn[n] = W_bn[n];
  __syncthreads();
  int i = blockIdx.x * 256 + tid;
  float xi = pos[i * 3], yi = pos[i * 3 + 1], zi = pos[i * 3 + 2];
  float h[CH];
#pragma unroll
  for (int c = 0; c < CH; ++c) h[c] = -__builtin_inff();
  for (int m = 0; m < K_NBR; ++m) {
    int j = nbr[i * K_TOT + 2 * m];  // dilation-2 slice of sorted top-32
    float f3 = pos[j * 3] - xi, f4 = pos[j * 3 + 1] - yi, f5 = pos[j * 3 + 2] - zi;
#pragma unroll
    for (int c = 0; c < CH; ++c) {
      float s = xi * s_Wp[0 * CH + c] + yi * s_Wp[1 * CH + c] +
                zi * s_Wp[2 * CH + c] + f3 * s_Wp[3 * CH + c] +
                f4 * s_Wp[4 * CH + c] + f5 * s_Wp[5 * CH + c];
      h[c] = fmaxf(h[c], s);  // leaky is monotone: apply after max
    }
  }
#pragma unroll
  for (int c = 0; c < CH; ++c) {
    float v = h[c] + b_pre[c];
    v = v > 0.f ? v : 0.2f * v;
    s_h[c * 256 + tid] = (_Float16)v;
  }
  float acc[CH_UP];
#pragma unroll
  for (int c2 = 0; c2 < CH_UP; ++c2) acc[c2] = b_bn[c2];
  for (int c = 0; c < CH; ++c) {
    float hv = (float)s_h[c * 256 + tid];
#pragma unroll
    for (int c2 = 0; c2 < CH_UP; ++c2) acc[c2] += hv * s_Wbn[c * CH_UP + c2];
  }
#pragma unroll
  for (int c2 = 0; c2 < CH_UP; ++c2) {
    float v = acc[c2];
    xf[i * CH_UP + c2] = v > 0.f ? v : 0.2f * v;
  }
}

// ---------------------------------------------------------------------------
// Kernel C: dilated EdgeConv 64->384 with WMMA. One wave per point: the 16
// messages are one 16-row A-tile; K=64 = 2 wmma steps; 24 column tiles.
// Fragment-0 is pure xf_i, fragment-1 is pure (xf_j - xf_i) -> built directly
// per lane, no LDS staging. Neighbor max = column-max of the C tile.
// ---------------------------------------------------------------------------
__global__ __launch_bounds__(256) void upconv_kernel(
    const float* __restrict__ xf, const int* __restrict__ nbr,
    const float* __restrict__ W, const float* __restrict__ b,
    float* __restrict__ res, int dil, int accumulate) {
  __shared__ alignas(32) _Float16 wfrag[2 * 24 * 512];  // 48 KB
  stage_w_frags(wfrag, W, 64, CHR, 24, 2 * 24, threadIdx.x, 256);
  __syncthreads();
  int wave = threadIdx.x >> 5, lane = threadIdx.x & 31;
  int point = blockIdx.x * 8 + wave;
  int m = lane & 15;  // message (neighbor) index = A row
  int j = nbr[point * K_TOT + m * dil];
  const float* xi = xf + point * CH_UP;
  const float* xj = xf + j * CH_UP;
  v16h a0, a1;
#pragma unroll
  for (int e = 0; e < 16; ++e) {
    int k = a_kmap(lane, e);
    float vi = xi[k];
    a0[e] = (_Float16)vi;             // features 0..31  = xf_i
    a1[e] = (_Float16)(xj[k] - vi);   // features 32..63 = xf_j - xf_i
  }
  for (int ct = 0; ct < 24; ++ct) {
    v16h b0 = *(const v16h*)(wfrag + ct * 512 + lane * 16);
    v16h b1 = *(const v16h*)(wfrag + (24 + ct) * 512 + lane * 16);
    v8f c = {};
    c = wmma_f16f32(a0, b0, c);
    c = wmma_f16f32(a1, b1, c);
    // max over the 16 message rows: rows split as 8 VGPRs x 2 lane-halves
    float mx = c[0];
#pragma unroll
    for (int v = 1; v < 8; ++v) mx = fmaxf(mx, c[v]);
    mx = fmaxf(mx, swap_half_wave(mx));
    if (lane < 16) {
      int ch = ct * 16 + lane;
      float val = mx + b[ch];
      val = val > 0.f ? val : 0.2f * val;
      float* dst = res + (size_t)point * CHR + ch;  // [N,384] == [4N,96]
      if (accumulate) *dst += val; else *dst = val;
    }
  }
}

// ---------------------------------------------------------------------------
// Kernel D1: t = [xf, pos] @ W_g1 + b_g1  (K=35 zero-padded to 64).
// Output written as [N,384] which is bit-identical to hg.reshape(4N,96).
// ---------------------------------------------------------------------------
__global__ __launch_bounds__(256) void g1_kernel(
    const float* __restrict__ xf, const float* __restrict__ pos,
    const float* __restrict__ W, const float* __restrict__ b,
    float* __restrict__ g) {
  __shared__ alignas(32) _Float16 wfrag[2 * 24 * 512];  // 48 KB
  stage_w_frags(wfrag, W, 35, CHR, 24, 2 * 24, threadIdx.x, 256);
  __syncthreads();
  int wave = threadIdx.x >> 5, lane = threadIdx.x & 31;
  int rt = blockIdx.x * 8 + wave;      // tile of 16 points
  int row = rt * 16 + (lane & 15);
  v16h a0, a1;
#pragma unroll
  for (int e = 0; e < 16; ++e) {
    int k = a_kmap(lane, e);
    a0[e] = (_Float16)xf[row * CH_UP + k];
    a1[e] = (_Float16)((k < 3) ? pos[row * 3 + k] : 0.f);  // K 32..34 = pos
  }
  for (int ct = 0; ct < 24; ++ct) {
    v16h b0 = *(const v16h*)(wfrag + ct * 512 + lane * 16);
    v16h b1 = *(const v16h*)(wfrag + (24 + ct) * 512 + lane * 16);
    v8f c = {};
    c = wmma_f16f32(a0, b0, c);
    c = wmma_f16f32(a1, b1, c);
    int ch = ct * 16 + (lane & 15);
    int rbase = rt * 16 + ((lane & 16) ? 8 : 0);
#pragma unroll
    for (int v = 0; v < 8; ++v)
      g[(size_t)(rbase + v) * CHR + ch] = c[v] + b[ch];
  }
}

// ---------------------------------------------------------------------------
// Kernel D2/E1: X[4N,96] @ W[96,96] (+bias), mode 0: out += (W_g2 path),
// mode 1: out = leaky_0.01(...) (W_r1 path, input pre-scaled by 1/3).
// ---------------------------------------------------------------------------
__global__ __launch_bounds__(256) void gemm96_kernel(
    const float* __restrict__ X, const float* __restrict__ W,
    const float* __restrict__ bias, float* __restrict__ out,
    float inscale, int mode) {
  __shared__ alignas(32) _Float16 wfrag[3 * 6 * 512];  // 18 KB
  stage_w_frags(wfrag, W, 96, 96, 6, 3 * 6, threadIdx.x, 256);
  __syncthreads();
  int wave = threadIdx.x >> 5, lane = threadIdx.x & 31;
  int rt = blockIdx.x * 8 + wave;
  size_t row = (size_t)rt * 16 + (lane & 15);
  v16h a0, a1, a2;
#pragma unroll
  for (int e = 0; e < 16; ++e) {
    int k = a_kmap(lane, e);
    a0[e] = (_Float16)(X[row * 96 + k] * inscale);
    a1[e] = (_Float16)(X[row * 96 + 32 + k] * inscale);
    a2[e] = (_Float16)(X[row * 96 + 64 + k] * inscale);
  }
  for (int ct = 0; ct < 6; ++ct) {
    v16h b0 = *(const v16h*)(wfrag + (0 * 6 + ct) * 512 + lane * 16);
    v16h b1 = *(const v16h*)(wfrag + (1 * 6 + ct) * 512 + lane * 16);
    v16h b2 = *(const v16h*)(wfrag + (2 * 6 + ct) * 512 + lane * 16);
    v8f c = {};
    c = wmma_f16f32(a0, b0, c);
    c = wmma_f16f32(a1, b1, c);
    c = wmma_f16f32(a2, b2, c);
    int ch = ct * 16 + (lane & 15);
    int rbase = rt * 16 + ((lane & 16) ? 8 : 0);
#pragma unroll
    for (int v = 0; v < 8; ++v) {
      float val = c[v] + bias[ch];
      float* dst = out + (size_t)(rbase + v) * 96 + ch;
      if (mode == 0) *dst += val;
      else           *dst = (val > 0.f) ? val : 0.01f * val;
    }
  }
}

// ---------------------------------------------------------------------------
// Kernel E2: q = u @ W_r2[96,3] + b_r2. Tiny N-dim -> plain VALU.
// ---------------------------------------------------------------------------
__global__ __launch_bounds__(256) void out_kernel(
    const float* __restrict__ u, const float* __restrict__ W,
    const float* __restrict__ b, float* __restrict__ q) {
  __shared__ float sW[96 * 3];
  int tid = threadIdx.x;
  for (int n = tid; n < 96 * 3; n += 256) sW[n] = W[n];
  __syncthreads();
  size_t row = (size_t)blockIdx.x * 256 + tid;
  float a0 = b[0], a1 = b[1], a2 = b[2];
  for (int c = 0; c < 96; ++c) {
    float v = u[row * 96 + c];
    a0 += v * sW[c * 3 + 0];
    a1 += v * sW[c * 3 + 1];
    a2 += v * sW[c * 3 + 2];
  }
  q[row * 3 + 0] = a0;
  q[row * 3 + 1] = a1;
  q[row * 3 + 2] = a2;
}

extern "C" void kernel_launch(void* const* d_in, const int* in_sizes, int n_in,
                              void* d_out, int out_size, void* d_ws,
                              size_t ws_size, hipStream_t stream) {
  const float* x     = (const float*)d_in[0];
  const float* W_pre = (const float*)d_in[1];
  const float* b_pre = (const float*)d_in[2];
  const float* W_bn  = (const float*)d_in[3];
  const float* b_bn  = (const float*)d_in[4];
  const float* W_up0 = (const float*)d_in[5];
  const float* b_up0 = (const float*)d_in[6];
  const float* W_up1 = (const float*)d_in[7];
  const float* b_up1 = (const float*)d_in[8];
  const float* W_g1  = (const float*)d_in[9];
  const float* b_g1  = (const float*)d_in[10];
  const float* W_g2  = (const float*)d_in[11];
  const float* b_g2  = (const float*)d_in[12];
  const float* W_r1  = (const float*)d_in[13];
  const float* b_r1  = (const float*)d_in[14];
  const float* W_r2  = (const float*)d_in[15];
  const float* b_r2  = (const float*)d_in[16];
  float* q = (float*)d_out;

  char* ws = (char*)d_ws;
  int*   nbr = (int*)ws;   ws += (size_t)N_PTS * K_TOT * sizeof(int);
  float* xf  = (float*)ws; ws += (size_t)N_PTS * CH_UP * sizeof(float);
  float* res = (float*)ws; ws += (size_t)NOUT * CH * sizeof(float);
  float* g   = (float*)ws;  // also reused as 'u' after D2 consumes it

  knn_kernel<<<N_PTS / 128, 128, 0, stream>>>(x, nbr);
  preconv_kernel<<<N_PTS / 256, 256, 0, stream>>>(x, nbr, W_pre, b_pre, W_bn,
                                                  b_bn, xf);
  // res  = EdgeConv(xf, dil=1); res += EdgeConv(xf, dil=2)
  upconv_kernel<<<N_PTS / 8, 256, 0, stream>>>(xf, nbr, W_up0, b_up0, res, 1, 0);
  upconv_kernel<<<N_PTS / 8, 256, 0, stream>>>(xf, nbr, W_up1, b_up1, res, 2, 1);
  // g    = [xf,pos] @ W_g1 + b_g1    (memory-layout == reshape(4N,96))
  g1_kernel<<<(N_PTS / 16) / 8, 256, 0, stream>>>(xf, x, W_g1, b_g1, g);
  // res += g @ W_g2 + b_g2
  gemm96_kernel<<<(NOUT / 16) / 8, 256, 0, stream>>>(g, W_g2, b_g2, res, 1.0f, 0);
  // g    = leaky_0.01((res/3) @ W_r1 + b_r1)
  gemm96_kernel<<<(NOUT / 16) / 8, 256, 0, stream>>>(res, W_r1, b_r1, g,
                                                     1.0f / 3.0f, 1);
  // q    = g @ W_r2 + b_r2
  out_kernel<<<NOUT / 256, 256, 0, stream>>>(g, W_r2, b_r2, q);
}
